// SimpleGNN_8254927142997
// MI455X (gfx1250) — compile-verified, matrix operand order checked by reading, not run on hardware
//
#include <hip/hip_runtime.h>
#include <hip/hip_bf16.h>
#include <math.h>

#define NN 50000
#define NE 800000
#define NG 512
#define DH 64
#define ED 16

typedef __attribute__((ext_vector_type(16))) __bf16 bf16x16;
typedef __attribute__((ext_vector_type(8)))  float  f32x8;

// ---------------------------------------------------------------------------
// Utility: zero a float buffer (grid-stride)
// ---------------------------------------------------------------------------
__global__ void zero_f32(float* __restrict__ p, int n) {
    for (int i = blockIdx.x * blockDim.x + threadIdx.x; i < n;
         i += gridDim.x * blockDim.x)
        p[i] = 0.0f;
}

// ---------------------------------------------------------------------------
// Convert weights to bf16 once per launch.
// We_bf: [4][32][64], K padded 16->32 with zeros.  Wh_bf: [3][64][64].
// ---------------------------------------------------------------------------
__global__ void convert_weights_kernel(const float* __restrict__ We,
                                       const float* __restrict__ Wh,
                                       __bf16* __restrict__ We_bf,
                                       __bf16* __restrict__ Wh_bf) {
    const int stride = gridDim.x * blockDim.x;
    const int t = blockIdx.x * blockDim.x + threadIdx.x;
    const int totWe = 4 * 32 * 64;
    for (int i = t; i < totWe; i += stride) {
        int l = i / (32 * 64);
        int r = (i / 64) % 32;
        int c = i % 64;
        We_bf[i] = (r < 16) ? (__bf16)We[(l * 16 + r) * 64 + c] : (__bf16)0.0f;
    }
    const int totWh = 3 * 64 * 64;
    for (int i = t; i < totWh; i += stride)
        Wh_bf[i] = (__bf16)Wh[i];
}

// ---------------------------------------------------------------------------
// Edge message + scatter. Persistent waves: weights (B operands, 4 col tiles)
// and bias live in registers; grid-stride over 16-edge tiles.
//   e    = edge_attr_tile[16x16] @ We[16x64]     (WMMA bf16, K padded to 32)
//   msg  = relu(h[src] + e + be)
//   aggr[dst] += msg                              (global_atomic_add_f32)
// ---------------------------------------------------------------------------
__global__ void __launch_bounds__(256) edge_msg_kernel(
        const float*  __restrict__ h,
        const float*  __restrict__ edge_attr,
        const int*    __restrict__ src,
        const int*    __restrict__ dst,
        const __bf16* __restrict__ We_bf,   // [32][64], rows 16..31 zero
        const float*  __restrict__ be,      // [64]
        float*        __restrict__ aggr) {
    const int lane   = threadIdx.x & 31;
    const int wave   = threadIdx.x >> 5;
    const int nWaves = gridDim.x * 8;
    const int m      = lane & 15;            // row (A) / col (B,C)
    const int hi     = lane >> 4;            // lane half
    const int kOff   = hi * 8;

    // ---- hoist B operands (whole 32x64 weight) + bias into registers ----
    bf16x16 B[4];
#pragma unroll
    for (int ct = 0; ct < 4; ++ct) {
#pragma unroll
        for (int v = 0; v < 4; ++v) {
            const int k0 = kOff + 2 * v;
            B[ct][2 * v + 0] = We_bf[(k0 + 0) * 64 + ct * 16 + m];
            B[ct][2 * v + 1] = We_bf[(k0 + 1) * 64 + ct * 16 + m];
        }
#pragma unroll
        for (int v = 4; v < 8; ++v) {
            const int k0 = 16 + kOff + 2 * (v - 4);
            B[ct][2 * v + 0] = We_bf[(k0 + 0) * 64 + ct * 16 + m];
            B[ct][2 * v + 1] = We_bf[(k0 + 1) * 64 + ct * 16 + m];
        }
    }
    float beh[4];
#pragma unroll
    for (int ct = 0; ct < 4; ++ct) beh[ct] = be[ct * 16 + m];

    for (int tile = blockIdx.x * 8 + wave; tile < NE / 16; tile += nWaves) {
        const int ebase = tile * 16;

        // one index load per lane; rows broadcast via shuffle later
        const int sm = src[ebase + m];
        const int dm = dst[ebase + m];

        // ---- A operand: 16x32 bf16, K=0..15 real, K=16..31 zero ---------
        const float* arow = edge_attr + (size_t)(ebase + m) * ED;
        // speculative prefetch of next grid-stride tile's row
        __builtin_prefetch(edge_attr + (size_t)(ebase + 16 * nWaves + m) * ED,
                           0, 1);
        bf16x16 A;
#pragma unroll
        for (int v = 0; v < 4; ++v) {
            A[2 * v + 0] = (__bf16)arow[kOff + 2 * v + 0];
            A[2 * v + 1] = (__bf16)arow[kOff + 2 * v + 1];
        }
#pragma unroll
        for (int v = 4; v < 8; ++v) {
            A[2 * v + 0] = (__bf16)0.0f;
            A[2 * v + 1] = (__bf16)0.0f;
        }

        f32x8 acc[4];
#pragma unroll
        for (int ct = 0; ct < 4; ++ct) {
            f32x8 c = {};
            acc[ct] = __builtin_amdgcn_wmma_f32_16x16x32_bf16(
                false, A, false, B[ct], (short)0, c, false, false);
        }

        // ---- gather h[src], relu, atomic scatter to aggr[dst] -----------
#pragma unroll
        for (int v = 0; v < 8; ++v) {
            const int row = v + 8 * hi;      // edge row in tile
            const int s = __shfl(sm, row, 32);
            const int d = __shfl(dm, row, 32);
            const float* hs = h + (size_t)s * DH;
            float*       ad = aggr + (size_t)d * DH;
#pragma unroll
            for (int ct = 0; ct < 4; ++ct) {
                const int col = ct * 16 + m;
                float msg = fmaxf(hs[col] + acc[ct][v] + beh[ct], 0.0f);
                atomicAdd(&ad[col], msg);
            }
        }
    }
}

// ---------------------------------------------------------------------------
// Node update (layers 0..2): h_out = relu((h_in + aggr) @ W + b)
// Persistent waves; full 64x64 weight held in registers (8 B operands).
// K=64 via two 16x16x32 WMMA steps per column tile.
// ---------------------------------------------------------------------------
__global__ void __launch_bounds__(256) node_update_kernel(
        const float*  __restrict__ h_in,
        const float*  __restrict__ aggr,
        const __bf16* __restrict__ W_bf,    // [64][64]
        const float*  __restrict__ bias,    // [64]
        float*        __restrict__ h_out) {
    const int lane   = threadIdx.x & 31;
    const int wave   = threadIdx.x >> 5;
    const int nWaves = gridDim.x * 8;
    const int m      = lane & 15;
    const int hi     = lane >> 4;

    // ---- hoist all 8 B operands (4 col tiles x 2 K-chunks) + bias -------
    bf16x16 B[4][2];
#pragma unroll
    for (int ct = 0; ct < 4; ++ct) {
#pragma unroll
        for (int kc = 0; kc < 2; ++kc) {
            const int kb = kc * 32 + hi * 8;
#pragma unroll
            for (int v = 0; v < 4; ++v) {
                const int k0 = kb + 2 * v;
                B[ct][kc][2 * v + 0] = W_bf[(k0 + 0) * 64 + ct * 16 + m];
                B[ct][kc][2 * v + 1] = W_bf[(k0 + 1) * 64 + ct * 16 + m];
            }
#pragma unroll
            for (int v = 4; v < 8; ++v) {
                const int k0 = kb + 16 + 2 * (v - 4);
                B[ct][kc][2 * v + 0] = W_bf[(k0 + 0) * 64 + ct * 16 + m];
                B[ct][kc][2 * v + 1] = W_bf[(k0 + 1) * 64 + ct * 16 + m];
            }
        }
    }
    float bh[4];
#pragma unroll
    for (int ct = 0; ct < 4; ++ct) bh[ct] = bias[ct * 16 + m];

    for (int tile = blockIdx.x * 8 + wave; tile < NN / 16; tile += nWaves) {
        const int nbase = tile * 16;

        // ---- A operands: z = h_in + aggr, bf16, two K-chunks ------------
        bf16x16 A[2];
        const float* hrow = h_in + (size_t)(nbase + m) * DH;
        const float* grow = aggr + (size_t)(nbase + m) * DH;
#pragma unroll
        for (int kc = 0; kc < 2; ++kc) {
            const int kb = kc * 32 + hi * 8;
#pragma unroll
            for (int v = 0; v < 4; ++v) {
                const int k0 = kb + 2 * v;
                A[kc][2 * v + 0] = (__bf16)(hrow[k0 + 0] + grow[k0 + 0]);
                A[kc][2 * v + 1] = (__bf16)(hrow[k0 + 1] + grow[k0 + 1]);
            }
#pragma unroll
            for (int v = 4; v < 8; ++v) {
                const int k0 = kb + 16 + 2 * (v - 4);
                A[kc][2 * v + 0] = (__bf16)(hrow[k0 + 0] + grow[k0 + 0]);
                A[kc][2 * v + 1] = (__bf16)(hrow[k0 + 1] + grow[k0 + 1]);
            }
        }

#pragma unroll
        for (int ct = 0; ct < 4; ++ct) {
            f32x8 acc = {};
            acc = __builtin_amdgcn_wmma_f32_16x16x32_bf16(
                false, A[0], false, B[ct][0], (short)0, acc, false, false);
            acc = __builtin_amdgcn_wmma_f32_16x16x32_bf16(
                false, A[1], false, B[ct][1], (short)0, acc, false, false);
#pragma unroll
            for (int v = 0; v < 8; ++v) {
                const int row = v + 8 * hi;
                const int col = ct * 16 + m;
                h_out[(size_t)(nbase + row) * DH + col] =
                    fmaxf(acc[v] + bh[ct], 0.0f);
            }
        }
    }
}

// ---------------------------------------------------------------------------
// Final layer: per-node (h+aggr)@W_out + b_out, pooled sums per graph.
// ---------------------------------------------------------------------------
__global__ void final_node_kernel(const float* __restrict__ h_in,
                                  const float* __restrict__ aggr,
                                  const float* __restrict__ W_out,  // [64]
                                  const float* __restrict__ b_out,  // [1]
                                  const int*   __restrict__ batch,
                                  float* __restrict__ pooled,
                                  float* __restrict__ counts) {
    const int n = blockIdx.x * blockDim.x + threadIdx.x;
    if (n >= NN) return;
    const float* hr = h_in + (size_t)n * DH;
    const float* ar = aggr + (size_t)n * DH;
    float acc = b_out[0];
#pragma unroll 8
    for (int k = 0; k < DH; ++k)
        acc += (hr[k] + ar[k]) * W_out[k];
    const int g = batch[n];
    atomicAdd(&pooled[g], acc);
    atomicAdd(&counts[g], 1.0f);
}

// ---------------------------------------------------------------------------
// Mean pool finalize + BatchNorm1d(1) over 512 graphs + sigmoid.
// ---------------------------------------------------------------------------
__global__ void bn_sigmoid_kernel(const float* __restrict__ pooled,
                                  const float* __restrict__ counts,
                                  const float* __restrict__ gamma,
                                  const float* __restrict__ beta,
                                  float* __restrict__ out) {
    __shared__ float red[NG];
    const int t = threadIdx.x;
    float p = pooled[t] / fmaxf(counts[t], 1.0f);

    red[t] = p;
    __syncthreads();
    for (int s = NG / 2; s > 0; s >>= 1) {
        if (t < s) red[t] += red[t + s];
        __syncthreads();
    }
    const float mean = red[0] / (float)NG;
    __syncthreads();

    const float d = p - mean;
    red[t] = d * d;
    __syncthreads();
    for (int s = NG / 2; s > 0; s >>= 1) {
        if (t < s) red[t] += red[t + s];
        __syncthreads();
    }
    const float var = red[0] / (float)NG;

    const float y = gamma[0] * d * rsqrtf(var + 1e-5f) + beta[0];
    out[t] = 1.0f / (1.0f + expf(-y));
}

// ---------------------------------------------------------------------------
extern "C" void kernel_launch(void* const* d_in, const int* in_sizes, int n_in,
                              void* d_out, int out_size, void* d_ws,
                              size_t ws_size, hipStream_t stream) {
    const float* x         = (const float*)d_in[0];
    const int*   eidx      = (const int*)d_in[1];    // [2, NE]
    const float* edge_attr = (const float*)d_in[2];
    const int*   batch     = (const int*)d_in[3];
    const float* W_hidden  = (const float*)d_in[4];  // [3,64,64]
    const float* b_hidden  = (const float*)d_in[5];  // [3,64]
    const float* We        = (const float*)d_in[6];  // [4,16,64]
    const float* be        = (const float*)d_in[7];  // [4,64]
    const float* W_out     = (const float*)d_in[8];  // [64]
    const float* b_out     = (const float*)d_in[9];
    const float* gamma     = (const float*)d_in[10];
    const float* beta      = (const float*)d_in[11];

    const int* srcp = eidx;
    const int* dstp = eidx + NE;

    // Workspace layout (~39 MB) -- hot data stays resident in the 192MB L2.
    char* ws = (char*)d_ws;
    float* hA = (float*)ws;      ws += sizeof(float) * (size_t)NN * DH;
    float* hB = (float*)ws;      ws += sizeof(float) * (size_t)NN * DH;
    float* aggr = (float*)ws;    ws += sizeof(float) * (size_t)NN * DH;
    __bf16* We_bf = (__bf16*)ws; ws += sizeof(__bf16) * 4 * 32 * 64;
    __bf16* Wh_bf = (__bf16*)ws; ws += sizeof(__bf16) * 3 * 64 * 64;
    float* pooled = (float*)ws;  ws += sizeof(float) * NG;
    float* counts = (float*)ws;  ws += sizeof(float) * NG;

    convert_weights_kernel<<<32, 256, 0, stream>>>(We, W_hidden, We_bf, Wh_bf);

    const int eblocks = 1024;    // persistent: 8192 waves, ~6 tiles each
    const int nblocks = 391;     // 3125 node tiles, ~1 per wave
    const int zn = NN * DH;

    const float* hcur = x;
    float* bufs[2] = {hA, hB};
    for (int l = 0; l < 3; ++l) {
        zero_f32<<<1024, 256, 0, stream>>>(aggr, zn);
        edge_msg_kernel<<<eblocks, 256, 0, stream>>>(
            hcur, edge_attr, srcp, dstp, We_bf + l * 32 * 64, be + l * DH, aggr);
        node_update_kernel<<<nblocks, 256, 0, stream>>>(
            hcur, aggr, Wh_bf + l * 64 * 64, b_hidden + l * DH, bufs[l & 1]);
        hcur = bufs[l & 1];
    }
    // Layer 3: aggregation + scalar projection + pooling
    zero_f32<<<1024, 256, 0, stream>>>(aggr, zn);
    edge_msg_kernel<<<eblocks, 256, 0, stream>>>(
        hcur, edge_attr, srcp, dstp, We_bf + 3 * 32 * 64, be + 3 * DH, aggr);
    zero_f32<<<2, 256, 0, stream>>>(pooled, NG);
    zero_f32<<<2, 256, 0, stream>>>(counts, NG);
    final_node_kernel<<<(NN + 255) / 256, 256, 0, stream>>>(
        hcur, aggr, W_out, b_out, batch, pooled, counts);
    bn_sigmoid_kernel<<<1, NG, 0, stream>>>(pooled, counts, gamma, beta,
                                            (float*)d_out);
}
